// RecurrentQNet_3152505996046
// MI455X (gfx1250) — compile-verified, matrix operand order checked by reading, not run on hardware
//
#include <hip/hip_runtime.h>
#include <hip/hip_bf16.h>

typedef __attribute__((ext_vector_type(16))) __bf16 v16bf;
typedef __attribute__((ext_vector_type(8)))  float  v8f;

#define B_  64
#define S_  1024
#define D_  64
#define H_  512
#define A_  16

// d_out layout (float elements): q | cb_preds | hidden | ca1_out
#define Q_OFF   ((size_t)0)
#define CB_OFF  ((size_t)1048576)   // B*S*A
#define HID_OFF ((size_t)5242880)   // + B*S*D
#define CA1_OFF ((size_t)5275648)   // + B*H

// ---------------------------------------------------------------------------
// Fragment swizzle: position of element (mn, k) inside a 16x32 (A) or 32x16
// (B) bf16 WMMA fragment tile, stored as 32 lanes x 16 elements contiguous.
// Per CDNA5 ISA 7.12.2: lanes 0-15 hold K 0-7/16-23, lanes 16-31 K 8-15/24-31,
// two bf16 per 32-bit reg.
// ---------------------------------------------------------------------------
__device__ __forceinline__ int frag_pos(int mn, int k) {
    int lane = mn + (((k >> 3) & 1) << 4);
    int i = ((k >> 4) << 3) + (((k >> 1) & 3) << 1) + (k & 1);
    return (lane << 4) + i;
}

__device__ __forceinline__ v16bf frag_ld(const __bf16* tile, int lane) {
    return ((const v16bf*)tile)[lane];
}

__device__ __forceinline__ v8f wmma_bf16(v16bf a, v16bf b, v8f c) {
    // D = A(16x32 bf16) * B(32x16 bf16) + C(16x16 f32)
    return __builtin_amdgcn_wmma_f32_16x16x32_bf16(
        /*neg_a=*/false, a, /*neg_b=*/false, b,
        /*c_mod=*/(short)0, c, /*reuse_a=*/false, /*reuse_b=*/false);
}

// ---------------------------------------------------------------------------
// Weight conversion: W[N][K] fp32 (torch layout, used as A @ W^T) -> bf16
// B-matrix fragments, tile layout [nt][kt], 512 bf16 per tile, pre-swizzled.
// ---------------------------------------------------------------------------
__global__ void wconv_kernel(const float* __restrict__ W, __bf16* __restrict__ dst,
                             int N, int K) {
    int idx = blockIdx.x * blockDim.x + threadIdx.x;
    if (idx >= N * K) return;
    int n = idx / K, k = idx - n * K;
    int nt = n >> 4, nl = n & 15, kt = k >> 5, kl = k & 31;
    int KT = K >> 5;
    dst[(((size_t)nt * KT + kt) << 9) + frag_pos(nl, kl)] = (__bf16)W[idx];
}

// ---------------------------------------------------------------------------
// Persistent single-workgroup GRU sequence kernel. 1024 threads = 32 waves on
// one WGP. Inputs, bf16 h and gates live in LDS (208 KB of the WGP's 320 KB); f32 h is in global h_ws.
// ---------------------------------------------------------------------------
__global__ __launch_bounds__(1024, 1)
void gru_seq_kernel(const float* __restrict__ x,
                    const float* __restrict__ b_ih, const float* __restrict__ b_hh,
                    const float* __restrict__ cb_b1, const float* __restrict__ cb_b2,
                    const __bf16* __restrict__ wih_f, const __bf16* __restrict__ whh_f,
                    const __bf16* __restrict__ cb1_f, const __bf16* __restrict__ cb2_f,
                    float* __restrict__ h_ws, __bf16* __restrict__ h_all,
                    float* __restrict__ out) {
    extern __shared__ char smem_raw[];
    __bf16* sInp = (__bf16*)smem_raw;   // inp fragments [mt<4][kt<4]   : 8192
    __bf16* sH   = sInp + 8192;         // h   fragments [mt<4][kt<16]  : 32768
    __bf16* sRT  = sH + 32768;          // r gate [b][c] / t1 fragments : 32768
    __bf16* sZ   = sRT + 32768;         // z gate [b][c]                : 32768

    const int tid  = threadIdx.x;
    const int lane = tid & 31;
    const int wv   = tid >> 5;
    const int mt   = wv & 3;       // batch row-tile owned by this wave
    const int ntB  = wv >> 2;      // base column-tile
    const int half = lane >> 4;
    const int nlo  = lane & 15;

    // ---- prologue: h0 = 0, pred0 = 0, load x[:,0,:] --------------------
    for (int i = tid; i < B_ * H_; i += 1024) h_ws[i] = 0.0f;
    for (int i = tid; i < 32768; i += 1024) sH[i] = (__bf16)0.0f;
    for (int i = tid; i < 8192;  i += 1024) sInp[i] = (__bf16)0.0f;
    for (int i = tid; i < B_ * D_; i += 1024) {
        int b = i >> 6, d = i & 63;
        sInp[(((b >> 4) << 2) + (d >> 5)) * 512 + frag_pos(b & 15, d & 31)] =
            (__bf16)x[((size_t)b * S_) * D_ + d];
    }
    __syncthreads();

    for (int t = 0; t < S_; ++t) {
        // ---- pass A: r/z gate tiles (gi+gh fused into one accumulator) ----
        for (int j = 0; j < 8; ++j) {
            int nt = ntB + 8 * j;                       // 0..63
            v8f acc = {};
            for (int kt = 0; kt < 4; ++kt)
                acc = wmma_bf16(frag_ld(sInp + (mt * 4 + kt) * 512, lane),
                                frag_ld(wih_f + ((size_t)(nt * 4 + kt) << 9), lane), acc);
            for (int kt = 0; kt < 16; ++kt)
                acc = wmma_bf16(frag_ld(sH + (mt * 16 + kt) * 512, lane),
                                frag_ld(whh_f + ((size_t)(nt * 16 + kt) << 9), lane), acc);
            int col = nt * 16 + nlo;                    // 0..1023
            float bias = b_ih[col] + b_hh[col];
            __bf16* dst = (nt < 32) ? sRT : sZ;
            int cc = (nt < 32) ? col : col - 512;
#pragma unroll
            for (int v = 0; v < 8; ++v) {
                int b = mt * 16 + v + 8 * half;
                float g = acc[v] + bias;
                dst[b * 512 + cc] = (__bf16)(1.0f / (1.0f + __expf(-g)));
            }
        }

        // ---- pass B: n-gate tiles, gi/gh kept separate ----
        v8f acci[4], acch[4];
        for (int j = 0; j < 4; ++j) {
            int nt = 64 + ntB + 8 * j;                  // 64..95
            v8f ai = {}, ah = {};
            for (int kt = 0; kt < 4; ++kt)
                ai = wmma_bf16(frag_ld(sInp + (mt * 4 + kt) * 512, lane),
                               frag_ld(wih_f + ((size_t)(nt * 4 + kt) << 9), lane), ai);
            for (int kt = 0; kt < 16; ++kt)
                ah = wmma_bf16(frag_ld(sH + (mt * 16 + kt) * 512, lane),
                               frag_ld(whh_f + ((size_t)(nt * 16 + kt) << 9), lane), ah);
            acci[j] = ai; acch[j] = ah;
        }
        __syncthreads();   // r/z visible; all reads of old sH complete

        // ---- finalize h_new = (1-z)*n + z*h ----
        for (int j = 0; j < 4; ++j) {
            int nt = 64 + ntB + 8 * j;
            int col = nt * 16 + nlo;                    // 1024..1535
            int ch = col - 1024;                        // 0..511
            float bi = b_ih[col], bh = b_hh[col];
#pragma unroll
            for (int v = 0; v < 8; ++v) {
                int b = mt * 16 + v + 8 * half;
                float r = (float)sRT[b * 512 + ch];
                float z = (float)sZ[b * 512 + ch];
                float n = tanhf(acci[j][v] + bi + r * (acch[j][v] + bh));
                float hold = h_ws[b * 512 + ch];
                float hn = (1.0f - z) * n + z * hold;
                h_ws[b * 512 + ch] = hn;
                h_all[((size_t)b * S_ + t) * H_ + ch] = (__bf16)hn;
                sH[(mt * 16 + (ch >> 5)) * 512 + frag_pos(b & 15, ch & 31)] = (__bf16)hn;
                if (t == S_ - 1) out[HID_OFF + (size_t)b * H_ + ch] = hn;
            }
        }
        __syncthreads();   // new sH complete

        // ---- callback layer 1: t1 = relu(h_new @ cb_w1^T + cb_b1) ----
        for (int j = 0; j < 4; ++j) {
            int nt1 = ntB + 8 * j;                      // 0..31
            v8f acc = {};
            for (int kt = 0; kt < 16; ++kt)
                acc = wmma_bf16(frag_ld(sH + (mt * 16 + kt) * 512, lane),
                                frag_ld(cb1_f + ((size_t)(nt1 * 16 + kt) << 9), lane), acc);
            int c1 = nt1 * 16 + nlo;
            float bb = cb_b1[c1];
#pragma unroll
            for (int v = 0; v < 8; ++v) {
                int ml = v + 8 * half;
                float tv = acc[v] + bb;
                tv = tv > 0.0f ? tv : 0.0f;
                sRT[(mt * 16 + (c1 >> 5)) * 512 + frag_pos(ml, c1 & 31)] = (__bf16)tv;
            }
        }
        __syncthreads();   // t1 fragments ready; sInp reads long done

        // ---- pred = t1 @ cb_w2^T + cb_b2 (waves 0..15); x[t+1] load (16..31)
        if (wv < 16) {
            int mtp = wv & 3, ntp = wv >> 2;
            v8f acc = {};
            for (int kt = 0; kt < 16; ++kt)
                acc = wmma_bf16(frag_ld(sRT + (mtp * 16 + kt) * 512, lane),
                                frag_ld(cb2_f + ((size_t)(ntp * 16 + kt) << 9), lane), acc);
            int cp = ntp * 16 + nlo;                    // 0..63
            float bb = cb_b2[cp];
#pragma unroll
            for (int v = 0; v < 8; ++v) {
                int ml = v + 8 * half;
                int b = mtp * 16 + ml;
                float p = acc[v] + bb;
                out[CB_OFF + ((size_t)b * S_ + t) * D_ + cp] = p;
                sInp[(mtp * 4 + 2 + (cp >> 5)) * 512 + frag_pos(b & 15, cp & 31)] = (__bf16)p;
            }
        } else if (t + 1 < S_) {
            int tid2 = tid - 512;                       // 0..511
            for (int e = 0; e < 8; ++e) {
                int i = tid2 * 8 + e;
                int b = i >> 6, d = i & 63;
                sInp[(((b >> 4) << 2) + (d >> 5)) * 512 + frag_pos(b & 15, d & 31)] =
                    (__bf16)x[((size_t)b * S_ + (t + 1)) * D_ + d];
            }
        }
        __syncthreads();   // sInp ready for step t+1
    }
}

// ---------------------------------------------------------------------------
// Head: ca1 = relu(out @ fc_w^T + fc_b); q = ca1 @ out_w^T + out_b.
// 4096 blocks, each owns 16 rows of the 65536-row activation matrix.
// ---------------------------------------------------------------------------
__global__ __launch_bounds__(256)
void head_kernel(const __bf16* __restrict__ h_all,
                 const __bf16* __restrict__ fc_f, const __bf16* __restrict__ outw_f,
                 const float* __restrict__ fc_b, const float* __restrict__ out_b,
                 float* __restrict__ out) {
    __shared__ __bf16 sC[16 * 512];     // ca1 row-tile as bf16 fragments [kt<16]
    const int tid = threadIdx.x, lane = tid & 31, wv = tid >> 5;
    const int half = lane >> 4, nlo = lane & 15;
    const size_t r0 = (size_t)blockIdx.x * 16;

    v8f zero = {};
    v8f acc[4] = {zero, zero, zero, zero};

    const int am = lane & 15;
    for (int kt = 0; kt < 16; ++kt) {
        // A fragment straight from row-major bf16 h_all (8x 32-bit pair loads)
        union { unsigned int u[8]; v16bf v; } af;
        const __bf16* base = h_all + (r0 + am) * H_ + kt * 32 + 8 * half;
#pragma unroll
        for (int i2 = 0; i2 < 8; ++i2) {
            int koff = ((i2 >> 2) << 4) + ((i2 & 3) << 1);
            af.u[i2] = *(const unsigned int*)(base + koff);
        }
#pragma unroll
        for (int j = 0; j < 4; ++j) {
            int nt = wv + 8 * j;
            acc[j] = wmma_bf16(af.v,
                               frag_ld(fc_f + ((size_t)(nt * 16 + kt) << 9), lane),
                               acc[j]);
        }
    }
#pragma unroll
    for (int j = 0; j < 4; ++j) {
        int nt = wv + 8 * j;
        int c = nt * 16 + nlo;
        float bb = fc_b[c];
#pragma unroll
        for (int v = 0; v < 8; ++v) {
            int ml = v + 8 * half;
            float val = acc[j][v] + bb;
            val = val > 0.0f ? val : 0.0f;
            out[CA1_OFF + (r0 + ml) * H_ + c] = val;
            sC[(c >> 5) * 512 + frag_pos(ml, c & 31)] = (__bf16)val;
        }
    }
    __syncthreads();

    if (wv == 0) {
        v8f q = {};
        for (int kt = 0; kt < 16; ++kt)
            q = wmma_bf16(frag_ld(sC + kt * 512, lane),
                          frag_ld(outw_f + ((size_t)kt << 9), lane), q);
        int c = nlo;                                    // A_=16, one col tile
        float bb = out_b[c];
#pragma unroll
        for (int v = 0; v < 8; ++v) {
            int ml = v + 8 * half;
            out[Q_OFF + (r0 + ml) * A_ + c] = q[v] + bb;
        }
    }
}

// ---------------------------------------------------------------------------
extern "C" void kernel_launch(void* const* d_in, const int* in_sizes, int n_in,
                              void* d_out, int out_size, void* d_ws, size_t ws_size,
                              hipStream_t stream) {
    const float* x     = (const float*)d_in[0];
    const float* W_ih  = (const float*)d_in[1];
    const float* W_hh  = (const float*)d_in[2];
    const float* b_ih  = (const float*)d_in[3];
    const float* b_hh  = (const float*)d_in[4];
    const float* fc_w  = (const float*)d_in[5];
    const float* fc_b  = (const float*)d_in[6];
    const float* out_w = (const float*)d_in[7];
    const float* out_b = (const float*)d_in[8];
    const float* cb_w1 = (const float*)d_in[9];
    const float* cb_b1 = (const float*)d_in[10];
    const float* cb_w2 = (const float*)d_in[11];
    const float* cb_b2 = (const float*)d_in[12];
    float* out = (float*)d_out;

    // workspace carve (bf16 elements)
    __bf16* wih_f  = (__bf16*)d_ws;            // 96*4*512   = 196608
    __bf16* whh_f  = wih_f  + 196608;          // 96*16*512  = 786432
    __bf16* cb1_f  = whh_f  + 786432;          // 32*16*512  = 262144
    __bf16* cb2_f  = cb1_f  + 262144;          // 4*16*512   = 32768
    __bf16* fc_f   = cb2_f  + 32768;           // 32*16*512  = 262144
    __bf16* outw_f = fc_f   + 262144;          // 1*16*512   = 8192
    __bf16* h_all  = outw_f + 8192;            // B*S*H      = 33554432
    float*  h_ws   = (float*)(h_all + 33554432); // B*H f32 state

    // 1) weight -> swizzled bf16 fragments
    wconv_kernel<<<(1536 * 128 + 255) / 256, 256, 0, stream>>>(W_ih, wih_f, 1536, 128);
    wconv_kernel<<<(1536 * 512 + 255) / 256, 256, 0, stream>>>(W_hh, whh_f, 1536, 512);
    wconv_kernel<<<(512 * 512 + 255) / 256, 256, 0, stream>>>(cb_w1, cb1_f, 512, 512);
    wconv_kernel<<<(64 * 512 + 255) / 256, 256, 0, stream>>>(cb_w2, cb2_f, 64, 512);
    wconv_kernel<<<(512 * 512 + 255) / 256, 256, 0, stream>>>(fc_w, fc_f, 512, 512);
    wconv_kernel<<<(16 * 512 + 255) / 256, 256, 0, stream>>>(out_w, outw_f, 16, 512);

    // 2) persistent recurrent kernel: 1 WGP, 32 waves, 208 KB LDS
    size_t shmem = (size_t)(8192 + 32768 + 32768 + 32768) * sizeof(__bf16);
    gru_seq_kernel<<<1, 1024, shmem, stream>>>(x, b_ih, b_hh, cb_b1, cb_b2,
                                               wih_f, whh_f, cb1_f, cb2_f,
                                               h_ws, h_all, out);

    // 3) feed-forward head over all B*S rows
    head_kernel<<<4096, 256, 0, stream>>>(h_all, fc_f, outw_f, fc_b, out_b, out);
}
